// Autoencoder_18468359372824
// MI455X (gfx1250) — matrix operand order repaired and checked by reading, not compiled, not run on hardware
//
#include <hip/hip_runtime.h>
#include <hip/hip_bf16.h>

// ---------------------------------------------------------------------------
// Problem constants (from reference): B=4, T=128, D=4096, HID=768, NC=32768,
// k = min(TOP_K*T, NC) = 8192.
// ---------------------------------------------------------------------------
#define BSZ   4
#define TOK   128
#define DIM   4096
#define HID   768
#define NC    32768
#define KSEL  8192
#define MROWS (BSZ * TOK)   // 512

typedef __bf16 bf16_t;
typedef __attribute__((ext_vector_type(16))) __bf16 v16bf;
typedef __attribute__((ext_vector_type(8)))  float  v8f;

// ---------------------------------------------------------------------------
// Generic bf16-WMMA GEMM:  C[M,N] = A[M,K] * op(B)  (+ bias*scale) (relu?)
//   BLAYOUT_NN == 0 : B is [N,K] row-major (compute A*B^T)  -- weight matrices
//   BLAYOUT_NN == 1 : B is [K,N] row-major (compute A*B)    -- dh matrix
// Block: 256 threads (8 waves). Block tile 128x64, wave tile 32x32, BK=64.
// All problem dims here divide the tile sizes exactly (no bounds checks).
// ---------------------------------------------------------------------------
template <typename TA, typename TB, int BLAYOUT_NN, int OUT_BF16, int RELU, int HAS_BIAS>
__global__ __launch_bounds__(256) void gemm_wmma_kernel(
    const TA* __restrict__ A, const TB* __restrict__ B,
    const float* __restrict__ bias, float bias_scale, void* __restrict__ Cout,
    int M, int N, int K,
    long long strideAb, long long strideBb, long long strideCb) {
  constexpr int BM = 128, BN = 64, BK = 64;
  __shared__ bf16_t As[BM * BK];   // 16 KB
  __shared__ bf16_t Bs[BN * BK];   //  8 KB

  const int tid  = threadIdx.x;
  const int lane = tid & 31;
  const int wave = tid >> 5;
  const int wr   = wave >> 1;   // 0..3 : 32-row block inside 128
  const int wc   = wave & 1;    // 0..1 : 32-col block inside 64

  const int m0 = blockIdx.y * BM;
  const int n0 = blockIdx.x * BN;
  const long long zA = (long long)blockIdx.z * strideAb;
  const long long zB = (long long)blockIdx.z * strideBb;
  const long long zC = (long long)blockIdx.z * strideCb;

  v8f acc[2][2];
#pragma unroll
  for (int i = 0; i < 2; ++i)
#pragma unroll
    for (int j = 0; j < 2; ++j)
#pragma unroll
      for (int r = 0; r < 8; ++r) acc[i][j][r] = 0.0f;

  for (int k0 = 0; k0 < K; k0 += BK) {
    // ---- stage A tile (BM x BK = 8192 elems, 32 per thread) ----
#pragma unroll
    for (int i = 0; i < 32; ++i) {
      int idx = tid + i * 256;
      int r = idx >> 6, c = idx & 63;
      As[r * BK + c] = (bf16_t)(float)A[zA + (long long)(m0 + r) * K + (k0 + c)];
    }
    // ---- stage B tile (BN x BK = 4096 elems, 16 per thread) ----
#pragma unroll
    for (int i = 0; i < 16; ++i) {
      int idx = tid + i * 256;
      if (BLAYOUT_NN) {  // B is [K,N]: Bs[n][k] = B[k0+kk][n0+n]
        int n = idx & 63, kk = idx >> 6;
        Bs[n * BK + kk] = (bf16_t)(float)B[zB + (long long)(k0 + kk) * N + (n0 + n)];
      } else {           // B is [N,K]: Bs[n][k] = B[n0+n][k0+k]
        int c = idx & 63, r = idx >> 6;
        Bs[r * BK + c] = (bf16_t)(float)B[zB + (long long)(n0 + r) * K + (k0 + c)];
      }
    }
    __syncthreads();

#pragma unroll
    for (int ks = 0; ks < BK; ks += 32) {
      // A fragment layout (16-bit A 16x32, ISA 7.12.2):
      //   lane<16: M=lane,   K in {kb+0..7, 16+kb+0..7} with kb=0
      //   lane>=16: M=lane-16, kb=8
      v16bf afrag[2], bfrag[2];
      const int ml = lane & 15;
      const int kbA = (lane >> 4) * 8;
#pragma unroll
      for (int i = 0; i < 2; ++i) {
        int row = wr * 32 + i * 16 + ml;
#pragma unroll
        for (int e = 0; e < 16; ++e) {
          int k = kbA + e + ((e >= 8) ? 8 : 0);
          afrag[i][e] = As[row * BK + ks + k];
        }
      }
      // B fragment (32x16): N=lane&15, K in {kb+0..7, 16+kb+0..7}, kb=(lane>>4)*8 (same as A)
      const int kbB = (lane >> 4) * 8;
#pragma unroll
      for (int j = 0; j < 2; ++j) {
        int col = wc * 32 + j * 16 + ml;
#pragma unroll
        for (int e = 0; e < 16; ++e) bfrag[j][e] = Bs[col * BK + ks + kbB + e + ((e >= 8) ? 8 : 0)];
      }
#pragma unroll
      for (int i = 0; i < 2; ++i)
#pragma unroll
        for (int j = 0; j < 2; ++j)
          acc[i][j] = __builtin_amdgcn_wmma_f32_16x16x32_bf16(
              false, afrag[i], false, bfrag[j], (short)0, acc[i][j], false, false);
    }
    __syncthreads();
  }

  // ---- epilogue: C/D layout: lane row-half via lane[4], col = lane&15 ----
  const int rowAdd = (lane >> 4) * 8;
  const int coll   = lane & 15;
#pragma unroll
  for (int i = 0; i < 2; ++i)
#pragma unroll
    for (int j = 0; j < 2; ++j)
#pragma unroll
      for (int r = 0; r < 8; ++r) {
        int row = m0 + wr * 32 + i * 16 + r + rowAdd;
        int col = n0 + wc * 32 + j * 16 + coll;
        float v = acc[i][j][r];
        if (HAS_BIAS) v += bias[col] * bias_scale;
        if (RELU) v = v > 0.0f ? v : 0.0f;
        long long off = zC + (long long)row * N + col;
        if (OUT_BF16) ((bf16_t*)Cout)[off] = (bf16_t)v;
        else          ((float*)Cout)[off]  = v;
      }
}

// ---------------------------------------------------------------------------
// logits_sum[b][n] = sum_t logits[b,t,n] / sqrt(T)
// ---------------------------------------------------------------------------
__global__ __launch_bounds__(256) void colsum_kernel(const bf16_t* __restrict__ logits,
                                                     float* __restrict__ sums) {
  int n = blockIdx.x * 256 + threadIdx.x;  // 0..NC-1
  int b = blockIdx.y;
  const bf16_t* p = logits + (long long)b * TOK * NC + n;
  float s = 0.0f;
  for (int t = 0; t < TOK; ++t) s += (float)p[(long long)t * NC];
  sums[b * NC + n] = s * 0.08838834764831845f;  // 1/sqrt(128)
}

// ---------------------------------------------------------------------------
// Per-batch full bitonic sort (descending by value, ascending index on ties)
// of all 32768 candidates in 256KB dynamic LDS; emit first 8192 indices.
// Matches jax.lax.top_k ordering (values sorted descending).
// ---------------------------------------------------------------------------
__global__ __launch_bounds__(1024) void topk_sort_kernel(const float* __restrict__ sums,
                                                         unsigned int* __restrict__ idx_out) {
  extern __shared__ unsigned long long keys[];  // 32768 * 8B = 256 KB
  const int b = blockIdx.x;
  const int tid = threadIdx.x;
  for (int i = tid; i < NC; i += 1024) {
    unsigned int u = __float_as_uint(sums[b * NC + i]);
    u = (u & 0x80000000u) ? ~u : (u | 0x80000000u);  // order-preserving map
    keys[i] = ((unsigned long long)u << 32) | (unsigned int)(~i);
  }
  __syncthreads();
  for (int ksz = 2; ksz <= NC; ksz <<= 1) {
    for (int j = ksz >> 1; j >= 1; j >>= 1) {
      for (int i = tid; i < NC; i += 1024) {
        int ixj = i ^ j;
        if (ixj > i) {
          unsigned long long a = keys[i], c = keys[ixj];
          bool desc = ((i & ksz) == 0);
          if (desc ? (a < c) : (a > c)) { keys[i] = c; keys[ixj] = a; }
        }
      }
      __syncthreads();
    }
  }
  for (int i = tid; i < KSEL; i += 1024)
    idx_out[b * KSEL + i] = ~((unsigned int)(keys[i] & 0xFFFFFFFFu));
}

// ---------------------------------------------------------------------------
// Gather components at idx, normalize, emit comps_k output, and compute
// dh[b,k,h] = relu(dec_w1[h,:] . comp + dec_b1[h]) stored bf16 [32768, 768].
// One wave per selected component row.
// ---------------------------------------------------------------------------
__global__ __launch_bounds__(256) void decode_comp_kernel(
    const unsigned int* __restrict__ idx, const float* __restrict__ components,
    const float* __restrict__ dec_w1, const float* __restrict__ dec_b1,
    float* __restrict__ comps_out, bf16_t* __restrict__ dh) {
  const int wave = threadIdx.x >> 5, lane = threadIdx.x & 31;
  const int row = blockIdx.x * 8 + wave;  // 0 .. BSZ*KSEL-1
  const unsigned int ci = idx[row];
  float c0 = components[ci * 3 + 0];
  float c1 = components[ci * 3 + 1];
  float c2 = components[ci * 3 + 2];
  float nrm = sqrtf(c0 * c0 + c1 * c1 + c2 * c2);
  float inv = 1.0f / fmaxf(nrm, 1e-12f);
  c0 *= inv; c1 *= inv; c2 *= inv;
  if (lane == 0) {
    comps_out[(long long)row * 3 + 0] = c0;
    comps_out[(long long)row * 3 + 1] = c1;
    comps_out[(long long)row * 3 + 2] = c2;
  }
  bf16_t* dhr = dh + (long long)row * HID;
#pragma unroll
  for (int i = 0; i < HID / 32; ++i) {
    int h = lane + i * 32;
    float v = dec_w1[h * 3 + 0] * c0 + dec_w1[h * 3 + 1] * c1 +
              dec_w1[h * 3 + 2] * c2 + dec_b1[h];
    dhr[h] = (bf16_t)(v > 0.0f ? v : 0.0f);
  }
}

// ---------------------------------------------------------------------------
// Gathered softmax: w[b,t,i] = softmax_i(logits[b,t,idx[b,i]]) * sqrt(k), bf16.
// One 256-thread block per (b,t) row; 32 gathered values per thread.
// ---------------------------------------------------------------------------
__global__ __launch_bounds__(256) void softmax_kernel(const bf16_t* __restrict__ logits,
                                                      const unsigned int* __restrict__ idx,
                                                      bf16_t* __restrict__ wout) {
  const int row = blockIdx.x;  // b*TOK + t
  const int b = row >> 7;
  const int tid = threadIdx.x;
  const bf16_t* lrow = logits + (long long)row * NC;
  const unsigned int* irow = idx + b * KSEL;
  float lv[32];
  float mx = -3.0e38f;
#pragma unroll
  for (int i = 0; i < 32; ++i) {
    float v = (float)lrow[irow[tid + i * 256]];
    lv[i] = v;
    mx = fmaxf(mx, v);
  }
  __shared__ float red[256];
  red[tid] = mx;
  __syncthreads();
  for (int s = 128; s > 0; s >>= 1) {
    if (tid < s) red[tid] = fmaxf(red[tid], red[tid + s]);
    __syncthreads();
  }
  mx = red[0];
  __syncthreads();
  float sum = 0.0f;
#pragma unroll
  for (int i = 0; i < 32; ++i) { lv[i] = __expf(lv[i] - mx); sum += lv[i]; }
  red[tid] = sum;
  __syncthreads();
  for (int s = 128; s > 0; s >>= 1) {
    if (tid < s) red[tid] += red[tid + s];
    __syncthreads();
  }
  const float scale = 90.50966799187809f / red[0];  // sqrt(8192)/sum
  bf16_t* wrow = wout + (long long)row * KSEL;
#pragma unroll
  for (int i = 0; i < 32; ++i) wrow[tid + i * 256] = (bf16_t)(lv[i] * scale);
}

// ---------------------------------------------------------------------------
// Orchestration. Key reassociation: x_recon = (W @ dh) @ dec_w2^T + sqrt(k)*b2
// (since softmax rows sum to 1), avoiding the 537MB dec_k tensor and cutting
// decoder FLOPs from ~240 GF to ~10 GF.
// ---------------------------------------------------------------------------
extern "C" void kernel_launch(void* const* d_in, const int* in_sizes, int n_in,
                              void* d_out, int out_size, void* d_ws, size_t ws_size,
                              hipStream_t stream) {
  (void)in_sizes; (void)n_in; (void)out_size; (void)ws_size;
  const float* x          = (const float*)d_in[0];
  const float* enc_w1     = (const float*)d_in[1];
  const float* enc_b1     = (const float*)d_in[2];
  const float* enc_w2     = (const float*)d_in[3];
  const float* enc_b2     = (const float*)d_in[4];
  const float* components = (const float*)d_in[5];
  const float* dec_w1     = (const float*)d_in[6];
  const float* dec_b1     = (const float*)d_in[7];
  const float* dec_w2     = (const float*)d_in[8];
  const float* dec_b2     = (const float*)d_in[9];

  float* x_recon   = (float*)d_out;                         // [512, 4096]
  float* comps_out = (float*)d_out + (long long)MROWS * DIM; // [4, 8192, 3]

  char* ws = (char*)d_ws;
  bf16_t*       hbuf   = (bf16_t*)(ws);                         // 512*768 bf16
  bf16_t*       logits = (bf16_t*)(ws + 786432);                // 512*32768 bf16
  float*        sums   = (float*)(ws + 34340864);               // 4*32768 f32
  unsigned int* idxbuf = (unsigned int*)(ws + 34865152);        // 4*8192 u32
  bf16_t*       dh     = (bf16_t*)(ws + 34996224);              // 32768*768 bf16
  bf16_t*       wbuf   = (bf16_t*)(ws + 85327872);              // 512*8192 bf16
  bf16_t*       ybuf   = (bf16_t*)(ws + 93716480);              // 512*768 bf16

  dim3 blk(256);

  // 1) h = relu(x @ enc_w1^T + enc_b1)        [512,768]
  gemm_wmma_kernel<float, float, 0, 1, 1, 1>
      <<<dim3(HID / 64, MROWS / 128, 1), blk, 0, stream>>>(
          x, enc_w1, enc_b1, 1.0f, hbuf, MROWS, HID, DIM, 0, 0, 0);

  // 2) logits = h @ enc_w2^T + enc_b2         [512,32768]
  gemm_wmma_kernel<bf16_t, float, 0, 1, 0, 1>
      <<<dim3(NC / 64, MROWS / 128, 1), blk, 0, stream>>>(
          hbuf, enc_w2, enc_b2, 1.0f, logits, MROWS, NC, HID, 0, 0, 0);

  // 3) logits_sum = sum_t logits / sqrt(T)    [4,32768]
  colsum_kernel<<<dim3(NC / 256, BSZ), blk, 0, stream>>>(logits, sums);

  // 4) per-batch descending sort -> top-8192 indices (sorted, jax order)
  topk_sort_kernel<<<dim3(BSZ), dim3(1024), (size_t)NC * 8, stream>>>(sums, idxbuf);

  // 5) gather+normalize components (output #2) and dh = relu(dec layer 1)
  decode_comp_kernel<<<dim3(BSZ * KSEL / 8), blk, 0, stream>>>(
      idxbuf, components, dec_w1, dec_b1, comps_out, dh);

  // 6) gathered softmax * sqrt(k)             [512,8192] bf16
  softmax_kernel<<<dim3(MROWS), blk, 0, stream>>>(logits, idxbuf, wbuf);

  // 7) Y = W @ dh  (batched over B)           [4][128,768]
  gemm_wmma_kernel<bf16_t, bf16_t, 1, 1, 0, 0>
      <<<dim3(HID / 64, TOK / 128, BSZ), blk, 0, stream>>>(
          wbuf, dh, nullptr, 0.0f, ybuf, TOK, HID, KSEL,
          (long long)TOK * KSEL, (long long)KSEL * HID, (long long)TOK * HID);

  // 8) x_recon = Y @ dec_w2^T + sqrt(k)*dec_b2  [512,4096] f32 -> output #1
  gemm_wmma_kernel<bf16_t, float, 0, 0, 0, 1>
      <<<dim3(DIM / 64, MROWS / 128, 1), blk, 0, stream>>>(
          ybuf, dec_w2, dec_b2, 90.50966799187809f, x_recon, MROWS, DIM, HID,
          0, 0, 0);
}
